// ShiftNetwork_70849780515433
// MI455X (gfx1250) — compile-verified, matrix operand order checked by reading, not run on hardware
//
#include <hip/hip_runtime.h>
#include <hip/hip_bf16.h>

typedef __attribute__((ext_vector_type(16))) _Float16 v16h;
typedef __attribute__((ext_vector_type(8)))  _Float16 v8h;
typedef __attribute__((ext_vector_type(8)))  float    v8f;

#define T_LEN   2048
#define SRATE   10
#define N_FULL  (T_LEN * SRATE)   // 20480
#define HID     1024
#define NKNOTS  32

// ---------------------------------------------------------------------------
// 1) p = basis_X @ B_knots / SCALE   (also second output)
// ---------------------------------------------------------------------------
__global__ __launch_bounds__(256) void compute_p_kernel(
    const float* __restrict__ basis_X, const float* __restrict__ B_knots,
    float* __restrict__ p_out) {
  int t = blockIdx.x * blockDim.x + threadIdx.x;
  if (t >= T_LEN) return;
  const float* row = basis_X + t * NKNOTS;
  float s = 0.f;
#pragma unroll
  for (int i = 0; i < NKNOTS; ++i) s += row[i] * B_knots[i];
  p_out[t] = s / 0.02f;
}

// ---------------------------------------------------------------------------
// 2) xs_full[n]: exact sparse evaluation of the P-matrix column sums.
//    P[t,n] = exp(-f(|n - p_exp[n] - 10 t| + 0.5)) is exactly 0.0f in fp32
//    unless the argument is within ~0.5 of 0 (15x repeated squaring / clamp).
//    Only t near round((n - p_exp)/10) can contribute; window +/-3 is ample.
// ---------------------------------------------------------------------------
__global__ __launch_bounds__(256) void compute_xs_kernel(
    const float* __restrict__ x, const float* __restrict__ p,
    float* __restrict__ xs_full) {
  int n = blockIdx.x * blockDim.x + threadIdx.x;
  if (n >= N_FULL) return;
  float pe = p[n / SRATE];
  float C = (float)n - pe;
  float psum = 0.f, xsum = 0.f;
  int t0 = (int)floorf(C * 0.1f + 0.5f);
#pragma unroll
  for (int dt = -3; dt <= 3; ++dt) {
    int t = t0 + dt;
    if (t < 0 || t >= T_LEN) continue;
    float d = C - (float)(t * SRATE);
    float pw = fabsf(d) + 0.5f;
#pragma unroll
    for (int it = 0; it < 15; ++it) pw = fminf(pw * pw, 1000.0f);
    float P = expf(-pw);
    psum += P;
    xsum += P * x[t];
  }
  float xs = (fabsf(psum) < 0.001f) ? __builtin_nanf("") : xsum;
  if (n == 0) xs = x[0];
  xs_full[n] = xs;
}

// ---------------------------------------------------------------------------
// 3) forward-fill (cummax-of-valid-index gather) + downsample by SRATE.
//    Strictly sequential recurrence over 20480 elements: trivial single-lane.
// ---------------------------------------------------------------------------
__global__ void ffill_kernel(const float* __restrict__ xs_full,
                             float* __restrict__ xs_ds) {
  float last = xs_full[0];  // index 0 forced valid (x[0])
  int di = 0;
  for (int n = 0; n < N_FULL; ++n) {
    float v = xs_full[n];
    if (!(v != v)) last = v;
    if (n % SRATE == 0) xs_ds[di++] = last;
  }
}

// ---------------------------------------------------------------------------
// 4) convert W2 / W3 to f16 once per call (workspace)
// ---------------------------------------------------------------------------
__global__ __launch_bounds__(256) void convert_w_kernel(
    const float* __restrict__ W2, const float* __restrict__ W3,
    _Float16* __restrict__ W2h, _Float16* __restrict__ W3h) {
  int i = blockIdx.x * blockDim.x + threadIdx.x;
  if (i < HID * HID) W2h[i] = (_Float16)W2[i];
  else {
    int j = i - HID * HID;
    W3h[j] = (_Float16)W3[j];
  }
}

// ---------------------------------------------------------------------------
// 5) layer1: outer product h1[t][j] = relu(xs[t]*W1[j] + b1[j]) -> f16
// ---------------------------------------------------------------------------
__global__ __launch_bounds__(256) void layer1_kernel(
    const float* __restrict__ xs, const float* __restrict__ W1,
    const float* __restrict__ b1, _Float16* __restrict__ H1h) {
  int i = blockIdx.x * blockDim.x + threadIdx.x;  // t*HID + j
  int j = i & (HID - 1);
  int t = i >> 10;
  float v = xs[t] * W1[j] + b1[j];
  H1h[i] = (_Float16)(v > 0.f ? v : 0.f);
}

// ---------------------------------------------------------------------------
// 6) WMMA GEMM: Out = relu(A @ W^T + bias), A: 2048x1024 f16, W: 1024x1024
//    row-major (out,in) so B-column j == W row j (contiguous K). One wave
//    computes a 16x64 strip (4 accumulators, A fragment reused 4x).
//    Fragment layout per CDNA5 ISA 7.12.2 (16-bit A 16x32):
//      lanes 0-15 : M=lane,    K = {k..k+7} in v0-3, {k+16..k+23} in v4-7
//      lanes 16-31: M=lane-16, K = {k+8..k+15},      {k+24..k+31}
//    B mirrors it with lane -> column. C/D: lane L, vgpr r -> (8*(L/16)+r, L%16).
// ---------------------------------------------------------------------------
__global__ __launch_bounds__(256) void gemm_relu_wmma_kernel(
    const _Float16* __restrict__ A, const _Float16* __restrict__ Wt,
    const float* __restrict__ bias, _Float16* __restrict__ Out) {
  const int K = HID;
  int wave = (blockIdx.x * blockDim.x + threadIdx.x) >> 5;  // 0..2047
  int lane = threadIdx.x & 31;
  int half = lane >> 4;    // selects K sub-block
  int l16 = lane & 15;
  int tM = wave >> 4;      // 0..127 : rows [tM*16, +16)
  int tN4 = wave & 15;     // 0..15  : cols [tN4*64, +64)

  const _Float16* Arow = A + (tM * 16 + l16) * K;
  const _Float16* Bbase = Wt + (tN4 * 64 + l16) * K;

  v8f zero = {0.f, 0.f, 0.f, 0.f, 0.f, 0.f, 0.f, 0.f};
  v8f acc0 = zero, acc1 = zero, acc2 = zero, acc3 = zero;

  for (int k = 0; k < K; k += 32) {
    v8h a_lo = *(const v8h*)(Arow + k + half * 8);
    v8h a_hi = *(const v8h*)(Arow + k + 16 + half * 8);
    v16h a = __builtin_shufflevector(a_lo, a_hi, 0, 1, 2, 3, 4, 5, 6, 7, 8, 9,
                                     10, 11, 12, 13, 14, 15);
#pragma unroll
    for (int nn = 0; nn < 4; ++nn) {
      const _Float16* Br = Bbase + nn * 16 * K;
      v8h b_lo = *(const v8h*)(Br + k + half * 8);
      v8h b_hi = *(const v8h*)(Br + k + 16 + half * 8);
      v16h b = __builtin_shufflevector(b_lo, b_hi, 0, 1, 2, 3, 4, 5, 6, 7, 8,
                                       9, 10, 11, 12, 13, 14, 15);
      v8f c = (nn == 0) ? acc0 : (nn == 1) ? acc1 : (nn == 2) ? acc2 : acc3;
      c = __builtin_amdgcn_wmma_f32_16x16x32_f16(false, a, false, b, (short)0,
                                                 c, false, false);
      if (nn == 0) acc0 = c; else if (nn == 1) acc1 = c;
      else if (nn == 2) acc2 = c; else acc3 = c;
    }
  }

#pragma unroll
  for (int nn = 0; nn < 4; ++nn) {
    v8f c = (nn == 0) ? acc0 : (nn == 1) ? acc1 : (nn == 2) ? acc2 : acc3;
    int col = tN4 * 64 + nn * 16 + l16;
    float bv = bias[col];
#pragma unroll
    for (int r = 0; r < 8; ++r) {
      int row = tM * 16 + half * 8 + r;
      float v = c[r] + bv;
      v = v > 0.f ? v : 0.f;
      Out[row * HID + col] = (_Float16)v;
    }
  }
}

// ---------------------------------------------------------------------------
// 7) layer4: y[t] = dot(h3[t], W4) + b4
// ---------------------------------------------------------------------------
__global__ __launch_bounds__(256) void layer4_kernel(
    const _Float16* __restrict__ H3, const float* __restrict__ W4,
    const float* __restrict__ b4, float* __restrict__ y) {
  int t = blockIdx.x * blockDim.x + threadIdx.x;
  if (t >= T_LEN) return;
  const _Float16* h = H3 + t * HID;
  float acc = 0.f;
  for (int j = 0; j < HID; ++j) acc += (float)h[j] * W4[j];
  y[t] = acc + b4[0];
}

extern "C" void kernel_launch(void* const* d_in, const int* in_sizes, int n_in,
                              void* d_out, int out_size, void* d_ws,
                              size_t ws_size, hipStream_t stream) {
  const float* x = (const float*)d_in[0];
  const float* basis_X = (const float*)d_in[1];
  const float* B_knots = (const float*)d_in[2];
  const float* W1 = (const float*)d_in[3];
  const float* b1 = (const float*)d_in[4];
  const float* W2 = (const float*)d_in[5];
  const float* b2 = (const float*)d_in[6];
  const float* W3 = (const float*)d_in[7];
  const float* b3 = (const float*)d_in[8];
  const float* W4 = (const float*)d_in[9];
  const float* b4 = (const float*)d_in[10];

  float* y_out = (float*)d_out;            // output 0: y (2048)
  float* p_out = (float*)d_out + T_LEN;    // output 1: p (2048)

  char* w = (char*)d_ws;
  float* xs_full = (float*)w;        w += N_FULL * sizeof(float);       // 80 KB
  float* xs_ds = (float*)w;          w += T_LEN * sizeof(float);        // 8 KB
  _Float16* W2h = (_Float16*)w;      w += HID * HID * sizeof(_Float16); // 2 MB
  _Float16* W3h = (_Float16*)w;      w += HID * HID * sizeof(_Float16); // 2 MB
  _Float16* H1h = (_Float16*)w;      w += T_LEN * HID * sizeof(_Float16);
  _Float16* H2h = (_Float16*)w;      w += T_LEN * HID * sizeof(_Float16);
  _Float16* H3h = (_Float16*)w;      w += T_LEN * HID * sizeof(_Float16);

  compute_p_kernel<<<(T_LEN + 255) / 256, 256, 0, stream>>>(basis_X, B_knots,
                                                            p_out);
  compute_xs_kernel<<<(N_FULL + 255) / 256, 256, 0, stream>>>(x, p_out,
                                                              xs_full);
  ffill_kernel<<<1, 1, 0, stream>>>(xs_full, xs_ds);
  convert_w_kernel<<<(2 * HID * HID) / 256, 256, 0, stream>>>(W2, W3, W2h,
                                                              W3h);
  layer1_kernel<<<(T_LEN * HID) / 256, 256, 0, stream>>>(xs_ds, W1, b1, H1h);
  // 2048 waves (128 M-tiles x 16 N-strips), 8 waves per block
  gemm_relu_wmma_kernel<<<256, 256, 0, stream>>>(H1h, W2h, b2, H2h);
  gemm_relu_wmma_kernel<<<256, 256, 0, stream>>>(H2h, W3h, b3, H3h);
  layer4_kernel<<<(T_LEN + 255) / 256, 256, 0, stream>>>(H3h, W4, b4, y_out);
}
